// SheafAttention_46909632807581
// MI455X (gfx1250) — compile-verified, matrix operand order checked by reading, not run on hardware
//
#include <hip/hip_runtime.h>
#include <hip/hip_bf16.h>

#define D_MODEL   1024
#define NUM_HEADS 16
#define BATCH     2
#define SEQ       2048
#define N_EDGES   32768
#define HEAD_DIM  64
#define MTOK      (BATCH * SEQ)          // 4096 tokens
#define MWORDS    (SEQ / 32)             // 64 mask words per row

typedef __attribute__((ext_vector_type(16))) __bf16 v16bf;
typedef __attribute__((ext_vector_type(8)))  __bf16 v8bf;
typedef __attribute__((ext_vector_type(8)))  float  v8f;
typedef unsigned u32x4 __attribute__((ext_vector_type(4)));
typedef int      i32x4 __attribute__((ext_vector_type(4)));
typedef int      i32x8 __attribute__((ext_vector_type(8)));

template<typename T> struct is_bf16_t { static constexpr bool value = false; };
template<>           struct is_bf16_t<__bf16> { static constexpr bool value = true; };

#if __has_builtin(__builtin_amdgcn_tensor_load_to_lds)
#define HAVE_TDM 1
#else
#define HAVE_TDM 0
#endif

// ---------- helpers ----------

__device__ __forceinline__ __bf16 f2bf(float f) {
  unsigned u = __builtin_bit_cast(unsigned, f);
  unsigned r = (u + 0x7fffu + ((u >> 16) & 1u)) >> 16;   // round-to-nearest-even
  unsigned short h = (unsigned short)r;
  return __builtin_bit_cast(__bf16, h);
}

__device__ __forceinline__ v16bf cat16(v8bf lo, v8bf hi) {
  return __builtin_shufflevector(lo, hi, 0,1,2,3,4,5,6,7,8,9,10,11,12,13,14,15);
}

__device__ __forceinline__ void stage8(const float* __restrict__ src, __bf16* dst) {
  float4 x0 = *(const float4*)src;
  float4 x1 = *(const float4*)(src + 4);
  dst[0] = f2bf(x0.x); dst[1] = f2bf(x0.y); dst[2] = f2bf(x0.z); dst[3] = f2bf(x0.w);
  dst[4] = f2bf(x1.x); dst[5] = f2bf(x1.y); dst[6] = f2bf(x1.z); dst[7] = f2bf(x1.w);
}

#if HAVE_TDM
// Issue a 2D TENSOR_LOAD_TO_LDS: tile (tile_d1 rows x tile_d0 elems) of 2-byte
// elements from a row-major tensor with row stride stride_d0, into LDS at
// lds_byte_off. D# layout per CDNA5 ISA ch.8 (group0: count/lds/global/type,
// group1: data_size/dims/tile/stride; groups 2-3 zero for <=2D).
// This toolchain's builtin takes 6 args (g0, g1, g2, g3, g4, cpol).
__device__ __forceinline__ void tdm_load_2d_b16(
    unsigned lds_byte_off, const void* gaddr,
    unsigned tile_d0, unsigned tile_d1,
    unsigned tensor_d0, unsigned tensor_d1, unsigned stride_d0)
{
  unsigned long long ga = (unsigned long long)gaddr;
  u32x4 g0;
  g0[0] = 1u;                                        // count=1 (valid user D#)
  g0[1] = lds_byte_off;                              // lds_addr
  g0[2] = (unsigned)(ga & 0xffffffffu);              // global_addr[31:0]
  g0[3] = (unsigned)((ga >> 32) & 0x01ffffffu)       // global_addr[56:32]
        | (2u << 30);                                // type=2 ("image")
  unsigned long long q0 = ((unsigned long long)(1u << 16))                 // data_size=2B
                        | ((unsigned long long)(tensor_d0 & 0xffffu) << 48);
  unsigned long long q1 = ((unsigned long long)(tensor_d0 >> 16) & 0xffffu)
                        | ((unsigned long long)tensor_d1 << 16)
                        | ((unsigned long long)(tile_d0 & 0xffffu) << 48);
  unsigned long long q2 = ((unsigned long long)(tile_d1 & 0xffffu))
                        | ((unsigned long long)stride_d0 << 32);           // stride[31:0]
  unsigned long long q3 = 0ull;                      // stride[47:32]=0, dim1_stride=0
  i32x8 g1;
  g1[0] = (int)(unsigned)(q0 & 0xffffffffu);
  g1[1] = (int)(unsigned)(q0 >> 32);
  g1[2] = (int)(unsigned)(q1 & 0xffffffffu);
  g1[3] = (int)(unsigned)(q1 >> 32);
  g1[4] = (int)(unsigned)(q2 & 0xffffffffu);
  g1[5] = (int)(unsigned)(q2 >> 32);
  g1[6] = (int)(unsigned)(q3 & 0xffffffffu);
  g1[7] = (int)(unsigned)(q3 >> 32);
  i32x4 z4 = (i32x4){0, 0, 0, 0};
  i32x8 z8 = (i32x8){0, 0, 0, 0, 0, 0, 0, 0};
  __builtin_amdgcn_tensor_load_to_lds(g0, g1, z4, z4, z8, 0);
}
#endif

// ---------- mask bitmask build ----------

__global__ void mask_init_kernel(unsigned* __restrict__ mb) {
  int idx = blockIdx.x * blockDim.x + threadIdx.x;      // SEQ * MWORDS words
  int row = idx >> 6;          // MWORDS == 64
  int w   = idx & 63;
  unsigned val = (w == (row >> 5)) ? (1u << (row & 31)) : 0u;
  mb[idx] = val;
}

__global__ void mask_edge_kernel(const int* __restrict__ e, unsigned* __restrict__ mb) {
  int t = blockIdx.x * blockDim.x + threadIdx.x;
  if (t >= N_EDGES) return;
  int i = e[2 * t], j = e[2 * t + 1];
  atomicOr(&mb[(size_t)i * MWORDS + (j >> 5)], 1u << (j & 31));
  atomicOr(&mb[(size_t)j * MWORDS + (i >> 5)], 1u << (i & 31));
}

// ---------- generic projection GEMM: out = A(4096x1024) @ W(1024x1024) + bias ----------
// 256 threads = 8 waves; block tile 128x64; K-step 32.
// Each wave owns 16 rows and reuses one A-fragment across 4 B-fragments (4 WMMAs).
// bf16 input path stages the A tile with global_load_async_to_lds_b128 (ASYNCcnt).

struct ProjLds {
  __bf16 A[128][48];    // 128x32 used; rows padded to 96 B for 16 B-aligned b128 LDS reads
  __bf16 Bt[64][48];    // B transposed: Bt[n][k]
};

template <typename TIn, bool TRANS_OUT, bool OUT_F32>
__global__ __launch_bounds__(256) void proj_kernel(
    const TIn* __restrict__ A, const float* __restrict__ W,
    const float* __restrict__ bias, void* __restrict__ outp)
{
  __shared__ ProjLds L;

  const int tid  = threadIdx.x;
  const int wave = tid >> 5;         // 0..7 -> 16-row slice
  const int lane = tid & 31;
  const int g    = lane >> 4;        // half-wave group (K-interleave select)
  const int ln   = lane & 15;
  const int mt = blockIdx.x >> 4;    // 32 m-tiles of 128
  const int nt = blockIdx.x & 15;    // 16 n-tiles of 64
  const int m0 = mt * 128, n0 = nt * 64;

  v8f acc[4];
  #pragma unroll
  for (int t = 0; t < 4; ++t) acc[t] = (v8f){0,0,0,0,0,0,0,0};

  const int a_m = tid >> 1;          // 0..127
  const int a_k = (tid & 1) << 4;    // 0 or 16 (each thread stages 16 halves)
  const int b_k = tid >> 3;          // 0..31
  const int b_n = (tid & 7) << 3;    // 0..56

  for (int k0 = 0; k0 < D_MODEL; k0 += 32) {
    // ---- stage A tile (128x32) into LDS as bf16 ----
    const TIn* asrc = A + (size_t)(m0 + a_m) * D_MODEL + k0 + a_k;
    if constexpr (is_bf16_t<TIn>::value) {
      // input already bf16: copy straight to LDS with the async data mover path
      unsigned dst0 = (unsigned)((a_m * 48 + a_k) * sizeof(__bf16));
      asm volatile("global_load_async_to_lds_b128 %0, %1, off"
                   :: "v"(dst0),        "v"(asrc)     : "memory");
      asm volatile("global_load_async_to_lds_b128 %0, %1, off"
                   :: "v"(dst0 + 16u),  "v"(asrc + 8) : "memory");
    } else {
      stage8(asrc,     &L.A[a_m][a_k]);
      stage8(asrc + 8, &L.A[a_m][a_k + 8]);
      if (k0 + 32 < D_MODEL)
        __builtin_prefetch(asrc + 32, 0, 1);
    }
    // ---- stage B^T tile: Bt[n][k] = W[k0+k][n0+n] (fp32 -> bf16) ----
    {
      const float* src = W + (size_t)(k0 + b_k) * D_MODEL + n0 + b_n;
      #pragma unroll
      for (int i = 0; i < 8; ++i)
        L.Bt[b_n + i][b_k] = f2bf(src[i]);
    }
    if constexpr (is_bf16_t<TIn>::value)
      asm volatile("s_wait_asynccnt 0x0" ::: "memory");
    __syncthreads();

    // ---- fragments + 4 WMMAs (A-fragment reused) ----
    const int mloc = wave * 16 + ln;
    v16bf afrag = cat16(*(const v8bf*)&L.A[mloc][8 * g],
                        *(const v8bf*)&L.A[mloc][16 + 8 * g]);
    #pragma unroll
    for (int t = 0; t < 4; ++t) {
      const int nb = t * 16 + ln;
      v16bf bfrag = cat16(*(const v8bf*)&L.Bt[nb][8 * g],
                          *(const v8bf*)&L.Bt[nb][16 + 8 * g]);
      acc[t] = __builtin_amdgcn_wmma_f32_16x16x32_bf16(false, afrag, false, bfrag,
                                                       (short)0, acc[t], false, false);
    }
    __syncthreads();
  }

  // epilogue: C layout -> element (m = v + 8g, n = ln) per 16x16 tile
  #pragma unroll
  for (int t = 0; t < 4; ++t) {
    #pragma unroll
    for (int v = 0; v < 8; ++v) {
      const int m  = m0 + wave * 16 + 8 * g + v;
      const int nn = n0 + t * 16 + ln;
      float r = acc[t][v] + bias[nn];
      if constexpr (OUT_F32) {
        ((float*)outp)[(size_t)m * D_MODEL + nn] = r;
      } else if constexpr (TRANS_OUT) {
        ((__bf16*)outp)[(size_t)nn * MTOK + m] = f2bf(r);
      } else {
        ((__bf16*)outp)[(size_t)m * D_MODEL + nn] = f2bf(r);
      }
    }
  }
}

// ---------- flash attention: one wave per (b, h, 16-query tile) ----------
// Q tile (16x64 bf16, reused across all key blocks) is brought in once via the
// Tensor Data Mover (TENSOR_LOAD_TO_LDS, TENSORcnt); P round-trips through LDS
// to convert C-layout -> A-layout for the PV WMMAs.

struct AttnLds {
  __bf16 Q[16][64];     // TDM destination at LDS offset 0 (rows contiguous, 128 B)
  __bf16 P[16][48];
};

__global__ __launch_bounds__(32) void sheaf_attn_kernel(
    const __bf16* __restrict__ Qb, const __bf16* __restrict__ Kb,
    const __bf16* __restrict__ Vt, const unsigned* __restrict__ maskbits,
    __bf16* __restrict__ Ao)
{
  __shared__ AttnLds SL;

  const int lane = threadIdx.x & 31;
  const int g = lane >> 4, ln = lane & 15;
  const int bx = blockIdx.x;
  const int qt = bx & 127;            // SEQ/16 = 128
  const int h  = (bx >> 7) & 15;
  const int b  = bx >> 11;
  const size_t tokBase = (size_t)b * SEQ;
  const int q0 = qt * 16;

  // Q fragments: 16 rows x hd=64 -> two K=32 chunks
  v16bf aq[2];
#if HAVE_TDM
  {
    tdm_load_2d_b16(/*lds_off=*/0u,
                    Qb + (tokBase + q0) * D_MODEL + h * HEAD_DIM,
                    /*tile_d0=*/HEAD_DIM, /*tile_d1=*/16,
                    /*tensor_d0=*/D_MODEL, /*tensor_d1=*/MTOK,
                    /*stride_d0=*/D_MODEL);
    __builtin_amdgcn_s_wait_tensorcnt(0);
    __syncthreads();
    const __bf16* qrow = &SL.Q[ln][0];
    #pragma unroll
    for (int c = 0; c < 2; ++c)
      aq[c] = cat16(*(const v8bf*)(qrow + c * 32 + 8 * g),
                    *(const v8bf*)(qrow + c * 32 + 16 + 8 * g));
  }
#else
  {
    const __bf16* qp = Qb + (tokBase + q0 + ln) * D_MODEL + h * HEAD_DIM;
    #pragma unroll
    for (int c = 0; c < 2; ++c)
      aq[c] = cat16(*(const v8bf*)(qp + c * 32 + 8 * g),
                    *(const v8bf*)(qp + c * 32 + 16 + 8 * g));
  }
#endif

  v8f o[4];
  v8f mrun, lrun;
  #pragma unroll
  for (int t = 0; t < 4; ++t) o[t] = (v8f){0,0,0,0,0,0,0,0};
  #pragma unroll
  for (int v = 0; v < 8; ++v) { mrun[v] = -1e30f; lrun[v] = 0.0f; }

  for (int j0 = 0; j0 < SEQ; j0 += 32) {
    // scores: two 16x16 tiles over this 32-key block
    v8f s[2];
    #pragma unroll
    for (int tn = 0; tn < 2; ++tn) {
      v8f acc = {0,0,0,0,0,0,0,0};
      const __bf16* kp = Kb + (tokBase + j0 + tn * 16 + ln) * D_MODEL + h * HEAD_DIM;
      #pragma unroll
      for (int c = 0; c < 2; ++c) {
        v16bf bf = cat16(*(const v8bf*)(kp + c * 32 + 8 * g),
                         *(const v8bf*)(kp + c * 32 + 16 + 8 * g));
        acc = __builtin_amdgcn_wmma_f32_16x16x32_bf16(false, aq[c], false, bf,
                                                      (short)0, acc, false, false);
      }
      s[tn] = acc;
    }

    // scale + topo mask (bitmask word covers exactly this 32-key block)
    const int wq = j0 >> 5;
    #pragma unroll
    for (int v = 0; v < 8; ++v) {
      const int iq = q0 + 8 * g + v;
      unsigned mw = maskbits[(size_t)iq * MWORDS + wq];
      float s0 = s[0][v] * 0.125f;     // 1/sqrt(64)
      float s1 = s[1][v] * 0.125f;
      s[0][v] = ((mw >> ln) & 1u)        ? s0 : -1e30f;
      s[1][v] = ((mw >> (16 + ln)) & 1u) ? s1 : -1e30f;
    }

    // online softmax (row reductions across the 16-lane half-wave)
    v8f alpha;
    #pragma unroll
    for (int v = 0; v < 8; ++v) {
      float mx = fmaxf(s[0][v], s[1][v]);
      #pragma unroll
      for (int off = 1; off < 16; off <<= 1)
        mx = fmaxf(mx, __shfl_xor(mx, off, 32));
      float mnew = fmaxf(mrun[v], mx);
      float al = __expf(mrun[v] - mnew);
      float p0 = __expf(s[0][v] - mnew);
      float p1 = __expf(s[1][v] - mnew);
      float rs = p0 + p1;
      #pragma unroll
      for (int off = 1; off < 16; off <<= 1)
        rs += __shfl_xor(rs, off, 32);
      lrun[v] = lrun[v] * al + rs;
      mrun[v] = mnew;
      alpha[v] = al;
      s[0][v] = p0; s[1][v] = p1;
    }

    // C-layout P -> LDS -> A-layout fragment (bf16)
    #pragma unroll
    for (int v = 0; v < 8; ++v) {
      SL.P[8 * g + v][ln]      = f2bf(s[0][v]);
      SL.P[8 * g + v][16 + ln] = f2bf(s[1][v]);
    }
    __syncthreads();
    v16bf pf = cat16(*(const v8bf*)&SL.P[ln][8 * g],
                     *(const v8bf*)&SL.P[ln][16 + 8 * g]);

    // O = O*alpha + P @ V  (V stored transposed: Vt[d][token])
    #pragma unroll
    for (int t = 0; t < 4; ++t) {
      const __bf16* vp = Vt + (size_t)(h * HEAD_DIM + t * 16 + ln) * MTOK + tokBase + j0;
      v16bf vf = cat16(*(const v8bf*)(vp + 8 * g),
                       *(const v8bf*)(vp + 16 + 8 * g));
      v8f oo = o[t];
      #pragma unroll
      for (int v = 0; v < 8; ++v) oo[v] *= alpha[v];
      o[t] = __builtin_amdgcn_wmma_f32_16x16x32_bf16(false, pf, false, vf,
                                                     (short)0, oo, false, false);
    }
    __syncthreads();
  }

  // finalize: divide by softmax denominator, store bf16 head slice
  #pragma unroll
  for (int t = 0; t < 4; ++t) {
    #pragma unroll
    for (int v = 0; v < 8; ++v) {
      float val = o[t][v] / lrun[v];
      size_t row = tokBase + q0 + 8 * g + v;
      Ao[row * D_MODEL + h * HEAD_DIM + t * 16 + ln] = f2bf(val);
    }
  }
}

// ---------- launch ----------

extern "C" void kernel_launch(void* const* d_in, const int* in_sizes, int n_in,
                              void* d_out, int out_size, void* d_ws, size_t ws_size,
                              hipStream_t stream) {
  const float* x  = (const float*)d_in[0];
  const float* Wq = (const float*)d_in[1];
  const float* bq = (const float*)d_in[2];
  const float* Wk = (const float*)d_in[3];
  const float* bk = (const float*)d_in[4];
  const float* Wv = (const float*)d_in[5];
  const float* bv = (const float*)d_in[6];
  const float* Wo = (const float*)d_in[7];
  const float* bo = (const float*)d_in[8];
  const int* edges = (const int*)d_in[9];
  float* out = (float*)d_out;

  char* ws = (char*)d_ws;
  const size_t MAT = (size_t)MTOK * D_MODEL * sizeof(__bf16);   // 8 MiB
  __bf16* Qb = (__bf16*)(ws);
  __bf16* Kb = (__bf16*)(ws + MAT);
  __bf16* Vt = (__bf16*)(ws + 2 * MAT);       // transposed: [D_MODEL][MTOK]
  __bf16* Ao = (__bf16*)(ws + 3 * MAT);
  unsigned* mb = (unsigned*)(ws + 4 * MAT);   // SEQ * MWORDS words = 512 KiB

  mask_init_kernel<<<(SEQ * MWORDS) / 256, 256, 0, stream>>>(mb);
  mask_edge_kernel<<<(N_EDGES + 255) / 256, 256, 0, stream>>>(edges, mb);

  dim3 pgrid((MTOK / 128) * (D_MODEL / 64));   // 512 blocks
  proj_kernel<float,  false, false><<<pgrid, 256, 0, stream>>>(x,  Wq, bq, (void*)Qb);
  proj_kernel<float,  false, false><<<pgrid, 256, 0, stream>>>(x,  Wk, bk, (void*)Kb);
  proj_kernel<float,  true,  false><<<pgrid, 256, 0, stream>>>(x,  Wv, bv, (void*)Vt);

  sheaf_attn_kernel<<<BATCH * NUM_HEADS * (SEQ / 16), 32, 0, stream>>>(Qb, Kb, Vt, mb, Ao);

  proj_kernel<__bf16, false, true ><<<pgrid, 256, 0, stream>>>(Ao, Wo, bo, (void*)out);
}
